// ProbSpareAttentionBlock_52828097741003
// MI455X (gfx1250) — compile-verified, matrix operand order checked by reading, not run on hardware
//
#include <hip/hip_runtime.h>
#include <math.h>

// ---------------------------------------------------------------------------
// ProbSparse attention transformer block for MI455X (gfx1250, wave32, WMMA).
// Dominant cost: 537MB attns output stream (HBM-bound, ~23us @ 23.3TB/s).
// GEMMs (21.5 GFLOP) run on v_wmma_f32_16x16x32_f16 with f32 accumulation.
// ---------------------------------------------------------------------------

#define B_   4
#define H_   8
#define LQ_  2048
#define LK_  2048
#define Dm_  512     // model dim = H*DK = H*DV
#define DK_  64
#define U_   32      // U_PART == U_TOP == 32
#define NROWS_ (B_*LQ_)   // 8192 rows for all (8192x512) GEMMs

typedef __attribute__((ext_vector_type(16))) _Float16 v16h;
typedef __attribute__((ext_vector_type(8)))  _Float16 v8h;
typedef __attribute__((ext_vector_type(8)))  float    v8f;

__device__ inline v8f v8f_zero() {
  v8f z = {0.f,0.f,0.f,0.f,0.f,0.f,0.f,0.f};
  return z;
}
__device__ inline float wave_sum(float v) {
  #pragma unroll
  for (int m = 16; m > 0; m >>= 1) v += __shfl_xor(v, m, 32);
  return v;
}
__device__ inline float wave_max(float v) {
  #pragma unroll
  for (int m = 16; m > 0; m >>= 1) v = fmaxf(v, __shfl_xor(v, m, 32));
  return v;
}
__device__ inline float gelu_exact(float x) {
  return 0.5f * x * (1.0f + erff(x * 0.70710678118654752f));
}

// ---------------------------------------------------------------------------
// Tiled WMMA GEMM: C[M,N] = act(A[M,K] @ W[K,N] + bias[N]); all f32 in/out,
// f16 operands staged in LDS, f32 accumulate. M%128==0, N%128==0, K%32==0.
// Block = 256 threads = 8 waves; wave (wm,wn) in 4x2 grid owns 32x64 subtile.
// ---------------------------------------------------------------------------
#define LDA_ 40   // f16 stride for 32-wide K tiles: 80 bytes, 16B aligned

template<int ACT>
__global__ void __launch_bounds__(256)
gemm_bias_kernel(const float* __restrict__ A, const float* __restrict__ W,
                 const float* __restrict__ bias, float* __restrict__ C,
                 int M, int N, int K)
{
  __shared__ __align__(16) _Float16 sA[128 * LDA_];
  __shared__ __align__(16) _Float16 sB[128 * LDA_];

  const int m0 = blockIdx.x * 128;
  const int n0 = blockIdx.y * 128;
  const int t  = threadIdx.x;
  const int wave = t >> 5, lane = t & 31;
  const int wm = wave >> 1, wn = wave & 1;       // 4x2 wave grid
  const int r  = lane & 15, hi = lane >> 4;

  v8f acc[2][4];
  #pragma unroll
  for (int s = 0; s < 2; ++s)
    #pragma unroll
    for (int u = 0; u < 4; ++u) acc[s][u] = v8f_zero();

  for (int kk = 0; kk < K; kk += 32) {
    // Stage A tile (128 x 32), f32 -> f16
    {
      const int row = t >> 1, half = t & 1;
      const float* src = A + (size_t)(m0 + row) * K + kk + half * 16;
      _Float16* dst = &sA[row * LDA_ + half * 16];
      #pragma unroll
      for (int j = 0; j < 16; ++j) dst[j] = (_Float16)src[j];
    }
    // Stage W tile transposed: W[kk+k][n0+n] -> sB[n][k]
    {
      const int k = t >> 3, nseg = t & 7;
      const float* src = W + (size_t)(kk + k) * N + n0 + nseg * 16;
      #pragma unroll
      for (int j = 0; j < 16; ++j) sB[(nseg * 16 + j) * LDA_ + k] = (_Float16)src[j];
    }
    __syncthreads();

    // A fragments: lane<16 holds row r, k {0..7,16..23}; lane>=16 k {8..15,24..31}
    v16h afrag[2], bfrag[4];
    #pragma unroll
    for (int s = 0; s < 2; ++s) {
      const _Float16* p = &sA[(wm * 32 + s * 16 + r) * LDA_];
      v8h lo  = *(const v8h*)(p + hi * 8);
      v8h hi8 = *(const v8h*)(p + 16 + hi * 8);
      #pragma unroll
      for (int j = 0; j < 8; ++j) { afrag[s][j] = lo[j]; afrag[s][8 + j] = hi8[j]; }
    }
    #pragma unroll
    for (int u = 0; u < 4; ++u) {
      const _Float16* p = &sB[(wn * 64 + u * 16 + r) * LDA_];
      v8h lo  = *(const v8h*)(p + hi * 8);
      v8h hi8 = *(const v8h*)(p + 16 + hi * 8);
      #pragma unroll
      for (int j = 0; j < 8; ++j) { bfrag[u][j] = lo[j]; bfrag[u][8 + j] = hi8[j]; }
    }
    #pragma unroll
    for (int s = 0; s < 2; ++s)
      #pragma unroll
      for (int u = 0; u < 4; ++u)
        acc[s][u] = __builtin_amdgcn_wmma_f32_16x16x32_f16(
            false, afrag[s], false, bfrag[u], (short)0, acc[s][u], false, false);
    __syncthreads();
  }

  // Epilogue: bias (+ GELU) + store (16x16 C layout: vgpr i, lane<16 -> M=i,
  // lane>=16 -> M=i+8; N = lane&15)
  #pragma unroll
  for (int s = 0; s < 2; ++s) {
    #pragma unroll
    for (int u = 0; u < 4; ++u) {
      const int col = n0 + wn * 64 + u * 16 + r;
      const float bv = bias[col];
      #pragma unroll
      for (int i = 0; i < 8; ++i) {
        const int row = m0 + wm * 32 + s * 16 + hi * 8 + i;
        float v = acc[s][u][i] + bv;
        if (ACT == 1) v = gelu_exact(v);
        C[(size_t)row * N + col] = v;
      }
    }
  }
}

// ---------------------------------------------------------------------------
// Sampled-score sparsity metric: M[b,h,l] = max_u(dot) - mean_u(dot), with a
// deterministic hash-based sample of U_=32 keys per query (the reference's
// _IDX_SAMPLE is a module-level RNG constant that is not an input).
// One wave32 per (b,h,l); lane u computes one sampled dot of length 64.
// ---------------------------------------------------------------------------
__global__ void __launch_bounds__(256)
sample_m_kernel(const float* __restrict__ Qp, const float* __restrict__ Kp,
                float* __restrict__ Mws)
{
  const int wid  = blockIdx.x * 8 + (threadIdx.x >> 5);   // 0 .. B*H*LQ-1
  const int lane = threadIdx.x & 31;
  const int l  = wid % LQ_;
  const int bh = wid / LQ_;
  const int b  = bh / H_, h = bh % H_;

  const unsigned idx =
      (2654435761u * (unsigned)l + 0x9E3779B9u * (unsigned)lane + 12345u) % (unsigned)LK_;
  const float* q = Qp + ((size_t)b * LQ_ + l) * Dm_ + h * DK_;
  const float* k = Kp + ((size_t)b * LK_ + idx) * Dm_ + h * DK_;
  float dot = 0.f;
  #pragma unroll 8
  for (int d = 0; d < DK_; ++d) dot += q[d] * k[d];

  const float mx = wave_max(dot);
  const float sm = wave_sum(dot);
  if (lane == 0) Mws[(size_t)bh * LQ_ + l] = mx - sm * (1.f / (float)U_);
}

// ---------------------------------------------------------------------------
// Top-32 indices per (b,h) by iterative argmax over 2048 entries (destroys Mws,
// which is fully recomputed each launch -> deterministic).
// ---------------------------------------------------------------------------
__global__ void __launch_bounds__(256)
topk_kernel(float* __restrict__ Mws, int* __restrict__ Mtop)
{
  const int bh = blockIdx.x;
  float* m = Mws + (size_t)bh * LQ_;
  __shared__ float sv[256];
  __shared__ int   si[256];
  const int t = threadIdx.x;

  for (int iter = 0; iter < U_; ++iter) {
    float bv = -INFINITY; int bi = 0;
    for (int i = t; i < LQ_; i += 256) {
      float v = m[i];
      if (v > bv) { bv = v; bi = i; }
    }
    sv[t] = bv; si[t] = bi;
    __syncthreads();
    for (int s = 128; s > 0; s >>= 1) {
      if (t < s && sv[t + s] > sv[t]) { sv[t] = sv[t + s]; si[t] = si[t + s]; }
      __syncthreads();
    }
    if (t == 0) { Mtop[bh * U_ + iter] = si[0]; m[si[0]] = -INFINITY; }
    __syncthreads();
  }
}

// ---------------------------------------------------------------------------
// V mean over keys: Vmean[b, h*64+d]  (laid out as (B, 512))
// ---------------------------------------------------------------------------
__global__ void __launch_bounds__(64)
vmean_kernel(const float* __restrict__ Vp, float* __restrict__ Vmean)
{
  const int bh = blockIdx.x;
  const int d  = threadIdx.x;
  const int b = bh / H_, h = bh % H_;
  float s = 0.f;
  for (int k = 0; k < LK_; ++k) s += Vp[((size_t)b * LK_ + k) * Dm_ + h * DK_ + d];
  Vmean[(size_t)b * Dm_ + h * DK_ + d] = s * (1.f / (float)LK_);
}

// ctx[b,l,h*64+d] = Vmean[b,h*64+d]  (broadcast), grid-stride
__global__ void __launch_bounds__(256)
fill_ctx_kernel(const float* __restrict__ Vmean, float* __restrict__ ctx)
{
  const size_t n = (size_t)B_ * LQ_ * Dm_;
  for (size_t e = blockIdx.x * (size_t)blockDim.x + threadIdx.x; e < n;
       e += (size_t)gridDim.x * blockDim.x) {
    const size_t col = e % Dm_;
    const size_t b   = e / ((size_t)LQ_ * Dm_);
    ctx[e] = Vmean[b * Dm_ + col];
  }
}

// attns default fill 1/LK: pure streaming float4 stores (the HBM bottleneck)
__global__ void __launch_bounds__(256)
fill_attns_kernel(float4* __restrict__ out, size_t n4, float val)
{
  const float4 v = {val, val, val, val};
  for (size_t i = blockIdx.x * (size_t)blockDim.x + threadIdx.x; i < n4;
       i += (size_t)gridDim.x * blockDim.x)
    out[i] = v;
}

// ---------------------------------------------------------------------------
// scores[bh, u, k] = (Q_red[u] . K[k]) / sqrt(DK) via WMMA.
// One block (8 waves) per (b,h). Q_red (32x64) staged to LDS as f16; each
// wave sweeps 16 of the 128 key-tiles, 2 q-tiles x 2 k-chunks of wmma.
// ---------------------------------------------------------------------------
__global__ void __launch_bounds__(256)
attn_scores_kernel(const float* __restrict__ Qp, const float* __restrict__ Kp,
                   const int* __restrict__ Mtop, float* __restrict__ scores)
{
  const int bh = blockIdx.x;
  const int b = bh / H_, h = bh % H_;
  __shared__ __align__(16) _Float16 sQ[32 * 72];   // 72 f16 stride = 144B

  const int t = threadIdx.x;
  {
    const int row = t >> 3, seg = t & 7;
    const int q = Mtop[bh * U_ + row];
    const float* src = Qp + ((size_t)b * LQ_ + q) * Dm_ + h * DK_ + seg * 8;
    _Float16* dst = &sQ[row * 72 + seg * 8];
    #pragma unroll
    for (int j = 0; j < 8; ++j) dst[j] = (_Float16)src[j];
  }
  __syncthreads();

  const int wave = t >> 5, lane = t & 31;
  const int r = lane & 15, hi = lane >> 4;

  v16h afrag[2][2];   // [q-subtile][k-chunk]
  #pragma unroll
  for (int s = 0; s < 2; ++s)
    #pragma unroll
    for (int c = 0; c < 2; ++c) {
      const _Float16* p = &sQ[(s * 16 + r) * 72 + c * 32];
      v8h lo  = *(const v8h*)(p + hi * 8);
      v8h hi8 = *(const v8h*)(p + 16 + hi * 8);
      #pragma unroll
      for (int j = 0; j < 8; ++j) { afrag[s][c][j] = lo[j]; afrag[s][c][8 + j] = hi8[j]; }
    }

  for (int kt = wave; kt < LK_ / 16; kt += 8) {
    const int k0 = kt * 16;
    v8f acc0 = v8f_zero(), acc1 = v8f_zero();
    #pragma unroll
    for (int c = 0; c < 2; ++c) {
      // B fragment: lane column = key k0+r; dims c*32 + {hi*8..+7, 16+hi*8..+7}
      const float* kp = Kp + ((size_t)b * LK_ + k0 + r) * Dm_ + h * DK_ + c * 32 + hi * 8;
      v16h bfrag;
      #pragma unroll
      for (int j = 0; j < 8; ++j) { bfrag[j] = (_Float16)kp[j]; bfrag[8 + j] = (_Float16)kp[16 + j]; }
      acc0 = __builtin_amdgcn_wmma_f32_16x16x32_f16(false, afrag[0][c], false, bfrag,
                                                    (short)0, acc0, false, false);
      acc1 = __builtin_amdgcn_wmma_f32_16x16x32_f16(false, afrag[1][c], false, bfrag,
                                                    (short)0, acc1, false, false);
    }
    const int key = k0 + r;
    #pragma unroll
    for (int i = 0; i < 8; ++i) {
      const int u0 = hi * 8 + i;
      scores[((size_t)bh * U_ + u0) * LK_ + key]      = acc0[i] * 0.125f;
      scores[((size_t)bh * U_ + 16 + u0) * LK_ + key] = acc1[i] * 0.125f;
    }
  }
}

// ---------------------------------------------------------------------------
// Softmax of one score row (2048) + scatter into attns[b,h,qidx,:]
// ---------------------------------------------------------------------------
__global__ void __launch_bounds__(256)
softmax_scatter_kernel(const float* __restrict__ scores, const int* __restrict__ Mtop,
                       float* __restrict__ attns)
{
  const int gid = blockIdx.x;            // bh*32 + u
  const int bh = gid >> 5, u = gid & 31;
  const int qidx = Mtop[bh * U_ + u];
  const float* row = scores + (size_t)gid * LK_;
  float* orow = attns + ((size_t)bh * LQ_ + qidx) * LK_;

  __shared__ float red[256];
  const int t = threadIdx.x;

  float vals[8];
  float lmax = -INFINITY;
  #pragma unroll
  for (int i = 0; i < 8; ++i) { vals[i] = row[t + i * 256]; lmax = fmaxf(lmax, vals[i]); }
  red[t] = lmax; __syncthreads();
  for (int s = 128; s > 0; s >>= 1) { if (t < s) red[t] = fmaxf(red[t], red[t + s]); __syncthreads(); }
  const float m = red[0]; __syncthreads();

  float lsum = 0.f;
  #pragma unroll
  for (int i = 0; i < 8; ++i) { vals[i] = expf(vals[i] - m); lsum += vals[i]; }
  red[t] = lsum; __syncthreads();
  for (int s = 128; s > 0; s >>= 1) { if (t < s) red[t] += red[t + s]; __syncthreads(); }
  const float inv = 1.f / red[0];
  #pragma unroll
  for (int i = 0; i < 8; ++i) orow[t + i * 256] = vals[i] * inv;
}

// ---------------------------------------------------------------------------
// ctx[b, qidx, h*64+d] = sum_k attn_row[k] * V[b,k,h*64+d]
// One wave per (bh,u); each lane owns 2 of the 64 output dims.
// ---------------------------------------------------------------------------
__global__ void __launch_bounds__(256)
upd_ctx_kernel(const float* __restrict__ attns, const float* __restrict__ Vp,
               const int* __restrict__ Mtop, float* __restrict__ ctx)
{
  const int wid  = blockIdx.x * 8 + (threadIdx.x >> 5);   // 0 .. 1023
  const int lane = threadIdx.x & 31;
  const int bh = wid >> 5, u = wid & 31;
  const int b = bh / H_, h = bh % H_;
  const int qidx = Mtop[bh * U_ + u];

  const float* arow = attns + ((size_t)bh * LQ_ + qidx) * LK_;
  const int d0 = lane * 2;
  float a0 = 0.f, a1 = 0.f;
  for (int k = 0; k < LK_; ++k) {
    const float w = arow[k];
    const float* vp = Vp + ((size_t)b * LK_ + k) * Dm_ + h * DK_ + d0;
    a0 += w * vp[0];
    a1 += w * vp[1];
  }
  float* dst = ctx + ((size_t)b * LQ_ + qidx) * Dm_ + h * DK_ + d0;
  dst[0] = a0; dst[1] = a1;
}

// ---------------------------------------------------------------------------
// out = LayerNorm(x + y) * gamma + beta ; one wave per 512-wide row
// ---------------------------------------------------------------------------
__global__ void __launch_bounds__(256)
add_ln_kernel(const float* __restrict__ x, const float* __restrict__ y,
              const float* __restrict__ gamma, const float* __restrict__ beta,
              float* __restrict__ out)
{
  const int row  = blockIdx.x * 8 + (threadIdx.x >> 5);
  const int lane = threadIdx.x & 31;
  const float* xr = x + (size_t)row * Dm_;
  const float* yr = y + (size_t)row * Dm_;

  float v[16]; float s = 0.f;
  #pragma unroll
  for (int i = 0; i < 16; ++i) { const int c = lane + i * 32; v[i] = xr[c] + yr[c]; s += v[i]; }
  const float mu = wave_sum(s) * (1.f / (float)Dm_);
  float var = 0.f;
  #pragma unroll
  for (int i = 0; i < 16; ++i) { const float d = v[i] - mu; var += d * d; }
  var = wave_sum(var) * (1.f / (float)Dm_);
  const float rinv = rsqrtf(var + 1e-6f);

  float* orow = out + (size_t)row * Dm_;
  #pragma unroll
  for (int i = 0; i < 16; ++i) {
    const int c = lane + i * 32;
    orow[c] = (v[i] - mu) * rinv * gamma[c] + beta[c];
  }
}

// ---------------------------------------------------------------------------
extern "C" void kernel_launch(void* const* d_in, const int* in_sizes, int n_in,
                              void* d_out, int out_size, void* d_ws, size_t ws_size,
                              hipStream_t stream)
{
  (void)in_sizes; (void)n_in; (void)out_size; (void)ws_size;

  const float* query = (const float*)d_in[0];
  const float* key   = (const float*)d_in[1];
  const float* value = (const float*)d_in[2];
  const float* Wq = (const float*)d_in[3];  const float* bq = (const float*)d_in[4];
  const float* Wk = (const float*)d_in[5];  const float* bk = (const float*)d_in[6];
  const float* Wv = (const float*)d_in[7];  const float* bv = (const float*)d_in[8];
  const float* Wo = (const float*)d_in[9];  const float* bo = (const float*)d_in[10];
  const float* Wf = (const float*)d_in[11]; const float* bf = (const float*)d_in[12];
  const float* gamma = (const float*)d_in[13];
  const float* beta  = (const float*)d_in[14];

  float* out_x = (float*)d_out;                               // (B,LQ,512)
  float* attns = out_x + (size_t)B_ * LQ_ * Dm_;              // (B,H,LQ,LK)

  // Workspace layout (floats); large buffers aliased after last use.
  const size_t MAT = (size_t)NROWS_ * Dm_;     // 4,194,304 floats = 16 MB
  float* p = (float*)d_ws;
  float* Qp       = p;  p += MAT;
  float* Kp       = p;  p += MAT;
  float* Vp       = p;  p += MAT;
  float* ctx      = p;  p += MAT;
  float* scoresWs = p;  p += (size_t)B_ * H_ * U_ * LK_;      // 8 MB
  float* Mws      = p;  p += (size_t)B_ * H_ * LQ_;
  float* Vmean    = p;  p += (size_t)B_ * Dm_;
  int*   Mtop     = (int*)p;                                   // B*H*32 ints
  float* attn_out = Kp;   // Kp dead after attn_scores_kernel
  float* x1       = Vp;   // Vp dead after upd_ctx_kernel
  float* ff       = Qp;   // Qp dead after attn_scores_kernel

  const dim3 gemm_grid(NROWS_ / 128, Dm_ / 128);   // 64 x 4

  // 1-3: Q/K/V projections (WMMA)
  gemm_bias_kernel<0><<<gemm_grid, 256, 0, stream>>>(query, Wq, bq, Qp, NROWS_, Dm_, Dm_);
  gemm_bias_kernel<0><<<gemm_grid, 256, 0, stream>>>(key,   Wk, bk, Kp, NROWS_, Dm_, Dm_);
  gemm_bias_kernel<0><<<gemm_grid, 256, 0, stream>>>(value, Wv, bv, Vp, NROWS_, Dm_, Dm_);

  // 4-5: sparsity metric + top-32 query selection per (b,h)
  sample_m_kernel<<<(B_ * H_ * LQ_) / 8, 256, 0, stream>>>(Qp, Kp, Mws);
  topk_kernel<<<B_ * H_, 256, 0, stream>>>(Mws, Mtop);

  // 6-8: default context = V-mean; default attns = 1/LK (HBM-bound stream)
  vmean_kernel<<<B_ * H_, 64, 0, stream>>>(Vp, Vmean);
  fill_ctx_kernel<<<2048, 256, 0, stream>>>(Vmean, ctx);
  const size_t n4 = ((size_t)B_ * H_ * LQ_ * LK_) / 4;
  fill_attns_kernel<<<8192, 256, 0, stream>>>((float4*)attns, n4, 1.0f / (float)LK_);

  // 9-11: reduced-query attention (WMMA scores -> softmax scatter -> ctx update)
  attn_scores_kernel<<<B_ * H_, 256, 0, stream>>>(Qp, Kp, Mtop, scoresWs);
  softmax_scatter_kernel<<<B_ * H_ * U_, 256, 0, stream>>>(scoresWs, Mtop, attns);
  upd_ctx_kernel<<<(B_ * H_ * U_) / 8, 256, 0, stream>>>(attns, Vp, Mtop, ctx);

  // 12-15: output projection, LN, FFN(+GELU), LN
  gemm_bias_kernel<0><<<gemm_grid, 256, 0, stream>>>(ctx, Wo, bo, attn_out, NROWS_, Dm_, Dm_);
  add_ln_kernel<<<NROWS_ / 8, 256, 0, stream>>>(query, attn_out, gamma, beta, x1);
  gemm_bias_kernel<1><<<gemm_grid, 256, 0, stream>>>(x1, Wf, bf, ff, NROWS_, Dm_, Dm_);
  add_ln_kernel<<<NROWS_ / 8, 256, 0, stream>>>(x1, ff, gamma, beta, out_x);
}
